// Custom_Matmul_75393855914400
// MI455X (gfx1250) — 8-bit matrix operand order repaired and checked by reading, compiled, not run on hardware
//
#include <hip/hip_runtime.h>

#define MDIM 8192
#define KDIM 4096
#define NDIM 4096
#define KSTEP 64
#define TILE_M 128
#define TILE_N 128
#define ROWB  68
#define SLABP (TILE_M * ROWB)

typedef __attribute__((ext_vector_type(8))) int      v8i;
typedef __attribute__((ext_vector_type(4))) int      v4i;
typedef __attribute__((ext_vector_type(2))) int      v2i;
typedef __attribute__((ext_vector_type(4))) unsigned u32x4;
typedef __attribute__((ext_vector_type(8))) int      i32x8;
typedef __attribute__((ext_vector_type(4))) int      i32x4;

__global__ __launch_bounds__(256) void pack_i32_to_i8(const int* __restrict__ src,
                                                      int* __restrict__ dst,
                                                      int n4) {
    int i = blockIdx.x * blockDim.x + threadIdx.x;
    if (i >= n4) return;
    v4i v = reinterpret_cast<const v4i*>(src)[i];
    dst[i] = (v.x & 0xFF) | ((v.y & 0xFF) << 8) |
             ((v.z & 0xFF) << 16) | (v.w << 24);
}

__device__ __forceinline__ void tdm_load_tile_2d(const void* gtile, unsigned ldsOff,
                                                 unsigned tileCols, unsigned tileRows,
                                                 unsigned tenD0, unsigned tenD1,
                                                 unsigned long long strideD0) {
    unsigned long long ga = (unsigned long long)(uintptr_t)gtile;
    u32x4 g0;
    g0.x = 1u;
    g0.y = ldsOff;
    g0.z = (unsigned)ga;
    g0.w = (unsigned)((ga >> 32) & 0x01FFFFFFu)
         | 0x80000000u;
    i32x8 g1;
    g1[0] = (int)((1u << 20) |
                  (3u << 22));
    g1[1] = (int)((tenD0 & 0xFFFFu) << 16);
    g1[2] = (int)(((tenD0 >> 16) & 0xFFFFu) |
                  ((tenD1 & 0xFFFFu) << 16));
    g1[3] = (int)(((tenD1 >> 16) & 0xFFFFu) |
                  ((tileCols & 0xFFFFu) << 16));
    g1[4] = (int)(tileRows & 0xFFFFu);
    g1[5] = (int)(unsigned)(strideD0 & 0xFFFFFFFFull);
    g1[6] = (int)(unsigned)((strideD0 >> 32) & 0xFFFFull);
    g1[7] = 0;
    i32x4 z4 = (i32x4){0, 0, 0, 0};
    i32x8 z8 = (i32x8){0, 0, 0, 0, 0, 0, 0, 0};
    __builtin_amdgcn_tensor_load_to_lds(g0, g1, z4, z4, z8, 0);
}

__device__ __forceinline__ v8i lds_a_frag(const char* p  ) {
    const v2i* q = reinterpret_cast<const v2i*>(p);
    v2i c0 = q[0], c1 = q[1], c2 = q[4], c3 = q[5];
    v8i a;
    a[0] = c0.x; a[1] = c0.y; a[2] = c1.x; a[3] = c1.y;
    a[4] = c2.x; a[5] = c2.y; a[6] = c3.x; a[7] = c3.y;
    return a;
}

__device__ __forceinline__ v8i lds_b_frag(const char* p  ) {
    const v4i* q = reinterpret_cast<const v4i*>(p);
    v4i lo = q[0], hi = q[2];
    v8i b;
    b[0] = lo.x; b[1] = lo.y; b[2] = lo.z; b[3] = lo.w;
    b[4] = hi.x; b[5] = hi.y; b[6] = hi.z; b[7] = hi.w;
    return b;
}

__device__ __forceinline__ v8i wmma_iu8(v8i a, v8i b, v8i c) {
    return __builtin_amdgcn_wmma_i32_16x16x64_iu8(true, a, true, b, c, false, false);
}

__global__ __launch_bounds__(256) void gemm_i8_wmma(
    const char* __restrict__ A8, const char* __restrict__ B8,
    const float* __restrict__ scale_A, const float* __restrict__ scale_B,
    const float* __restrict__ scale_O, float* __restrict__ out) {

    __shared__ __attribute__((aligned(16))) char smem[4 * SLABP];

    const int lane = threadIdx.x & 31;
    const int wave = threadIdx.x >> 5;
    const int wm = wave & 3;
    const int wn = wave >> 2;
    const int l15 = lane & 15;
    const int hi  = lane >> 4;

    const int mBase = blockIdx.y * TILE_M + wm * 32;
    const int nBase = blockIdx.x * TILE_N + wn * 64;

    const char* aTileG = A8 + (size_t)(blockIdx.y * TILE_M) * KDIM;
    const char* bTileG = B8 + (size_t)(blockIdx.x * TILE_N) * KDIM;

    const unsigned smemBase = (unsigned)(uintptr_t)(&smem[0]);

    const char* aFragBase = smem + (wm * 32 + l15) * ROWB + hi * 16;
    const char* bFragBase = smem + 2 * SLABP + (wn * 64 + l15) * ROWB + hi * 16;

    v8i c00 = {}, c01 = {}, c02 = {}, c03 = {};
    v8i c10 = {}, c11 = {}, c12 = {}, c13 = {};

    const int NK = KDIM / KSTEP;

    if (wave == 0) {
        tdm_load_tile_2d(aTileG, smemBase, KSTEP, TILE_M, KDIM, MDIM, KDIM);
        tdm_load_tile_2d(bTileG, smemBase + 2 * SLABP, KSTEP, TILE_N, KDIM, NDIM, KDIM);
    }

    for (int i = 0; i < NK; ++i) {
        const int cur = i & 1;
        if (wave == 0) {
            if (i + 1 < NK) {
                const int nxt = cur ^ 1;
                const int kn = (i + 1) * KSTEP;
                tdm_load_tile_2d(aTileG + kn, smemBase + nxt * SLABP,
                                 KSTEP, TILE_M, KDIM, MDIM, KDIM);
                tdm_load_tile_2d(bTileG + kn, smemBase + 2 * SLABP + nxt * SLABP,
                                 KSTEP, TILE_N, KDIM, NDIM, KDIM);
                __builtin_amdgcn_s_wait_tensorcnt(2);
            } else {
                __builtin_amdgcn_s_wait_tensorcnt(0);
            }
        }
        __syncthreads();

        const char* aS = aFragBase + cur * SLABP;
        const char* bS = bFragBase + cur * SLABP;
        v8i a0 = lds_a_frag(aS);
        v8i a1 = lds_a_frag(aS + 16 * ROWB);
        v8i b0 = lds_b_frag(bS);
        v8i b1 = lds_b_frag(bS + 16 * ROWB);
        v8i b2 = lds_b_frag(bS + 32 * ROWB);
        v8i b3 = lds_b_frag(bS + 48 * ROWB);

        c00 = wmma_iu8(a0, b0, c00);
        c01 = wmma_iu8(a0, b1, c01);
        c02 = wmma_iu8(a0, b2, c02);
        c03 = wmma_iu8(a0, b3, c03);
        c10 = wmma_iu8(a1, b0, c10);
        c11 = wmma_iu8(a1, b1, c11);
        c12 = wmma_iu8(a1, b2, c12);
        c13 = wmma_iu8(a1, b3, c13);

        __syncthreads();
    }

    const v8i accArr[2][4] = {{c00, c01, c02, c03}, {c10, c11, c12, c13}};
    const int mOff = hi * 8;
    #pragma unroll
    for (int ti = 0; ti < 2; ++ti) {
        #pragma unroll
        for (int tj = 0; tj < 4; ++tj) {
            const int n = nBase + tj * 16 + l15;
            const float sb = scale_B[n];
            const int m0 = mBase + ti * 16 + mOff;
            #pragma unroll
            for (int r = 0; r < 8; ++r) {
                const int m = m0 + r;
                const float s = scale_A[m] * sb / scale_O[m];
                float x = rintf((float)accArr[ti][tj][r] * s);
                x = fminf(127.0f, fmaxf(-128.0f, x));
                out[(size_t)m * NDIM + n] = x;
            }
        }
    }
}

__global__ __launch_bounds__(256) void copy_scales(const float* __restrict__ s,
                                                   float* __restrict__ dst, int n) {
    int i = blockIdx.x * blockDim.x + threadIdx.x;
    if (i < n) dst[i] = s[i];
}

extern "C" void kernel_launch(void* const* d_in, const int* in_sizes, int n_in,
                              void* d_out, int out_size, void* d_ws, size_t ws_size,
                              hipStream_t stream) {
    const int*   A  = (const int*)d_in[0];
    const float* sA = (const float*)d_in[1];
    const int*   B  = (const int*)d_in[2];
    const float* sB = (const float*)d_in[3];
    const float* sO = (const float*)d_in[4];
    float* out = (float*)d_out;

    char* A8 = (char*)d_ws;
    char* B8 = A8 + (size_t)MDIM * KDIM;

    const int nA4 = (MDIM * KDIM) / 4;
    const int nB4 = (NDIM * KDIM) / 4;
    pack_i32_to_i8<<<nA4 / 256, 256, 0, stream>>>(A, (int*)A8, nA4);
    pack_i32_to_i8<<<nB4 / 256, 256, 0, stream>>>(B, (int*)B8, nB4);

    dim3 grid(NDIM / TILE_N, MDIM / TILE_M);
    gemm_i8_wmma<<<grid, 256, 0, stream>>>(A8, B8, sA, sB, sO, out);

    copy_scales<<<MDIM / 256, 256, 0, stream>>>(sO, out + (size_t)MDIM * NDIM, MDIM);
}
